// KGlayer_25056839204916
// MI455X (gfx1250) — matrix operand order repaired and checked by reading, not compiled, not run on hardware
//
#include <hip/hip_runtime.h>

#define LATDIM 128
#define WAVES  4   // waves (16-row tiles) per 128-thread block

typedef __attribute__((ext_vector_type(16))) __bf16 v16bf;
typedef __attribute__((ext_vector_type(8)))  float  v8f;

// ---------------------------------------------------------------------------
// Weight pack: fp32 [K=128, N=128] row-major -> bf16 WMMA B-fragment layout.
// Packed index p = ((kb*8 + nt)*32 + lane)*16 + e
//   lane in 0..31: n = (lane&15) + nt*16 ; K = kb*32 + (lane>>4)*8 + (e&7) + (e>>3)*16
// so each lane later loads 16 contiguous bf16 (32B) per (kb, nt).
// ---------------------------------------------------------------------------
__global__ void pack_weights_kernel(const float* __restrict__ qT,
                                    const float* __restrict__ kT,
                                    const float* __restrict__ vT,
                                    __bf16* __restrict__ pack) {
    const float* src = (blockIdx.x == 0) ? qT : (blockIdx.x == 1 ? kT : vT);
    __bf16* dst = pack + (size_t)blockIdx.x * (128 * 128);
    for (int p = threadIdx.x; p < 128 * 128; p += blockDim.x) {
        int e    = p & 15;
        int lane = (p >> 4) & 31;
        int nt   = (p >> 9) & 7;
        int kb   = p >> 12;
        int n = (lane & 15) + nt * 16;
        int K = kb * 32 + (lane >> 4) * 8 + (e & 7) + (e >> 3) * 16;
        dst[p] = (__bf16)src[K * 128 + n];
    }
}

__global__ void zero_kernel(float* __restrict__ p, long long n) {
    long long i = (long long)blockIdx.x * blockDim.x + threadIdx.x;
    long long stride = (long long)gridDim.x * blockDim.x;
    for (; i < n; i += stride) p[i] = 0.0f;
}

// ---------------------------------------------------------------------------
// One wave computes C[16 x 128] = A[16 x 128] (fp32, row-major, cvt->bf16)
//                               x B[128 x 128] (pre-packed bf16 fragments)
// acc[nt] is the 16x16 f32 C tile for columns nt*16..nt*16+15.
// A-fragment per ISA layout: lane m = l&15; kbase = (l>>4)*8;
//   elements 0..7  -> K = kb*32 + kbase + 0..7
//   elements 8..15 -> K = kb*32 + kbase + 16..23
// ---------------------------------------------------------------------------
static __device__ __forceinline__ void gemm16x128(const float* __restrict__ A,
                                                  long long row0, long long rowMax,
                                                  const __bf16* __restrict__ bpack,
                                                  v8f acc[8]) {
    const int lane  = threadIdx.x & 31;
    const int m     = lane & 15;
    const int kbase = (lane >> 4) * 8;
    long long r = row0 + m;
    if (r > rowMax) r = rowMax;
    const float* arow = A + r * LATDIM;

#pragma unroll
    for (int kb = 0; kb < 4; ++kb) {
        const float* p = arow + kb * 32 + kbase;
        float4 a0 = *(const float4*)(p);
        float4 a1 = *(const float4*)(p + 4);
        float4 a2 = *(const float4*)(p + 16);
        float4 a3 = *(const float4*)(p + 20);
        v16bf a;
        a[0]  = (__bf16)a0.x; a[1]  = (__bf16)a0.y; a[2]  = (__bf16)a0.z; a[3]  = (__bf16)a0.w;
        a[4]  = (__bf16)a1.x; a[5]  = (__bf16)a1.y; a[6]  = (__bf16)a1.z; a[7]  = (__bf16)a1.w;
        a[8]  = (__bf16)a2.x; a[9]  = (__bf16)a2.y; a[10] = (__bf16)a2.z; a[11] = (__bf16)a2.w;
        a[12] = (__bf16)a3.x; a[13] = (__bf16)a3.y; a[14] = (__bf16)a3.z; a[15] = (__bf16)a3.w;

#pragma unroll
        for (int nt = 0; nt < 8; ++nt) {
            v16bf b = *(const v16bf*)(bpack + (((kb * 8 + nt) * 32 + lane) << 4));
            acc[nt] = __builtin_amdgcn_wmma_f32_16x16x32_bf16(
                false, a, false, b, (short)0, acc[nt], false, false);
        }
    }
}

// ---------------------------------------------------------------------------
// Kernel: nodeQ[N,128] = embeds @ qTrans   (one wave per 16-row tile)
// C layout: lane l, acc[nt][r] -> row = r + 8*(l>>4), col = (l&15) + nt*16
// ---------------------------------------------------------------------------
__global__ void nodeq_kernel(const float* __restrict__ embeds,
                             const __bf16* __restrict__ qpack,
                             float* __restrict__ nodeQ, int N) {
    const int wave = threadIdx.x >> 5;
    const int lane = threadIdx.x & 31;
    long long nTiles = (N + 15) / 16;
    long long tile = (long long)blockIdx.x * WAVES + wave;
    if (tile >= nTiles) return;

    v8f acc[8] = {};
    gemm16x128(embeds, tile * 16, (long long)N - 1, qpack, acc);

    const int half = lane >> 4, n0 = lane & 15;
#pragma unroll
    for (int nt = 0; nt < 8; ++nt)
#pragma unroll
        for (int r = 0; r < 8; ++r) {
            long long rowi = tile * 16 + r + 8 * half;
            if (rowi < N) nodeQ[rowi * LATDIM + n0 + nt * 16] = acc[nt][r];
        }
}

// ---------------------------------------------------------------------------
// Pass A: k = colEmbeds @ kTrans (WMMA -> LDS); att = exp(clip(q.k)) per head;
//         expAtt stored; attNorm accumulated with f32 atomics.
// ---------------------------------------------------------------------------
__global__ void att_kernel(const int* __restrict__ rows,
                           const float* __restrict__ colEmbeds,
                           const float* __restrict__ nodeQ,
                           const __bf16* __restrict__ kpack,
                           float* __restrict__ expAtt,
                           float* __restrict__ attNorm, int E) {
    __shared__ float kS[WAVES][16 * LATDIM];
    __shared__ int   rowsS[WAVES][16];

    const int wave = threadIdx.x >> 5;
    const int lane = threadIdx.x & 31;
    long long nTiles = (E + 15) / 16;
    long long tile = (long long)blockIdx.x * WAVES + wave;
    const bool active = tile < nTiles;
    const long long e0 = tile * 16;

    if (active) {
        if (lane < 16) {
            long long e = e0 + lane; if (e >= E) e = E - 1;
            rowsS[wave][lane] = rows[e];
        }
        v8f acc[8] = {};
        gemm16x128(colEmbeds, e0, (long long)E - 1, kpack, acc);
        const int half = lane >> 4, n0 = lane & 15;
#pragma unroll
        for (int nt = 0; nt < 8; ++nt)
#pragma unroll
            for (int r = 0; r < 8; ++r)
                kS[wave][(r + 8 * half) * LATDIM + n0 + nt * 16] = acc[nt][r];
    }
    __syncthreads();

    if (active) {
        const int m = lane & 15, half = lane >> 4;
        long long e = e0 + m;
        if (e < E) {
            int row = rowsS[wave][m];
            const float* q = nodeQ + (long long)row * LATDIM;
            const float* k = &kS[wave][m * LATDIM];
#pragma unroll
            for (int hh = 0; hh < 2; ++hh) {
                int h = half * 2 + hh;
                float dot = 0.0f;
#pragma unroll
                for (int j = 0; j < 32; ++j)
                    dot = fmaf(q[h * 32 + j], k[h * 32 + j], dot);
                dot = fminf(fmaxf(dot, -10.0f), 10.0f);
                float ex = expf(dot);
                expAtt[e * 4 + h] = ex;
                atomicAdd(&attNorm[(long long)row * 4 + h], ex);
            }
        }
    }
}

// ---------------------------------------------------------------------------
// Pass B: v = colEmbeds @ vTrans (WMMA -> LDS); res = att*v scattered into
//         out[rows] with coalesced f32 atomics.
// ---------------------------------------------------------------------------
__global__ void out_kernel(const int* __restrict__ rows,
                           const float* __restrict__ colEmbeds,
                           const __bf16* __restrict__ vpack,
                           const float* __restrict__ expAtt,
                           const float* __restrict__ attNorm,
                           float* __restrict__ out, int E) {
    __shared__ float vS[WAVES][16 * LATDIM];
    __shared__ int   rowsS[WAVES][16];
    __shared__ float attS[WAVES][64];

    const int wave = threadIdx.x >> 5;
    const int lane = threadIdx.x & 31;
    long long nTiles = (E + 15) / 16;
    long long tile = (long long)blockIdx.x * WAVES + wave;
    const bool active = tile < nTiles;
    const long long e0 = tile * 16;

    if (active) {
        if (lane < 16) {
            long long e = e0 + lane; if (e >= E) e = E - 1;
            rowsS[wave][lane] = rows[e];
        }
        v8f acc[8] = {};
        gemm16x128(colEmbeds, e0, (long long)E - 1, vpack, acc);
        const int half = lane >> 4, n0 = lane & 15;
#pragma unroll
        for (int nt = 0; nt < 8; ++nt)
#pragma unroll
            for (int r = 0; r < 8; ++r)
                vS[wave][(r + 8 * half) * LATDIM + n0 + nt * 16] = acc[nt][r];

        // normalized attention weights: 64 values, 2 per lane
#pragma unroll
        for (int t = 0; t < 2; ++t) {
            int idx = lane * 2 + t;          // 0..63
            int m = idx >> 2, h = idx & 3;
            long long e = e0 + m; if (e >= E) e = E - 1;
            int row = rows[e];
            float ex = expAtt[e * 4 + h];
            float nrm = attNorm[(long long)row * 4 + h];
            attS[wave][idx] = ex / (nrm + 1e-8f);
        }
    }
    __syncthreads();

    if (active) {
        for (int idx = lane; idx < 16 * LATDIM; idx += 32) {
            int m = idx >> 7;
            int d = idx & 127;
            long long e = e0 + m;
            if (e < E) {
                int row = rowsS[wave][m];
                float val = attS[wave][m * 4 + (d >> 5)] * vS[wave][m * LATDIM + d];
                atomicAdd(&out[(long long)row * LATDIM + d], val);
            }
        }
    }
}

// ---------------------------------------------------------------------------
extern "C" void kernel_launch(void* const* d_in, const int* in_sizes, int n_in,
                              void* d_out, int out_size, void* d_ws, size_t ws_size,
                              hipStream_t stream) {
    const int*   rows      = (const int*)d_in[0];
    const float* colEmbeds = (const float*)d_in[1];
    const float* embeds    = (const float*)d_in[2];
    const float* qT        = (const float*)d_in[3];
    const float* kT        = (const float*)d_in[4];
    const float* vT        = (const float*)d_in[5];
    const int E = in_sizes[0];
    const int N = in_sizes[2] / LATDIM;
    float* out = (float*)d_out;

    // workspace layout
    char* ws = (char*)d_ws;
    __bf16* pack = (__bf16*)ws;                                  // 3 x 32 KB
    size_t off = (size_t)3 * 128 * 128 * sizeof(__bf16);
    off = (off + 255) & ~(size_t)255;
    float* attNorm = (float*)(ws + off); off += (size_t)N * 4 * sizeof(float);
    off = (off + 255) & ~(size_t)255;
    float* expAtt  = (float*)(ws + off); off += (size_t)E * 4 * sizeof(float);
    off = (off + 255) & ~(size_t)255;
    float* nodeQ   = (float*)(ws + off);

    pack_weights_kernel<<<3, 256, 0, stream>>>(qT, kT, vT, pack);
    zero_kernel<<<1024, 256, 0, stream>>>(attNorm, (long long)N * 4);
    zero_kernel<<<2048, 256, 0, stream>>>(out, (long long)N * LATDIM);

    long long nTilesN = (N + 15) / 16;
    int blocksN = (int)((nTilesN + WAVES - 1) / WAVES);
    nodeq_kernel<<<blocksN, WAVES * 32, 0, stream>>>(embeds, pack, nodeQ, N);

    long long nTilesE = (E + 15) / 16;
    int blocksE = (int)((nTilesE + WAVES - 1) / WAVES);
    att_kernel<<<blocksE, WAVES * 32, 0, stream>>>(rows, colEmbeds, nodeQ,
                                                   pack + 128 * 128, expAtt, attNorm, E);
    out_kernel<<<blocksE, WAVES * 32, 0, stream>>>(rows, colEmbeds,
                                                   pack + 2 * 128 * 128, expAtt, attNorm,
                                                   out, E);
}
